// DeformableAttention_67156108640967
// MI455X (gfx1250) — matrix operand order repaired and checked by reading, not compiled, not run on hardware
//
#include <hip/hip_runtime.h>
#include <hip/hip_bf16.h>

typedef __bf16 bf16_t;
typedef __attribute__((ext_vector_type(16))) __bf16 v16bf;
typedef __attribute__((ext_vector_type(8)))  __bf16 v8bf;
typedef __attribute__((ext_vector_type(8)))  float  v8f;

#define BATCH 4
#define CCH   256
#define HH    64
#define WW    64
#define HWSZ  4096
#define NGRP  4
#define CG    64
#define NHEAD 8
#define HC    32
#define NSAMP 256
#define RPEW  127

// ---------------------------------------------------------------------------
// A-tile loader: 16x32 bf16, row-major source with leading dim 256.
// Contiguous k order, matching load_b32x16: lane<16 holds row m=lane, K=0..15; lane>=16 holds
// row m=lane-16, K=16..31. Both runs are 16B aligned -> vector loads.
__device__ __forceinline__ v16bf load_a16x32(const bf16_t* tile, int lane) {
  int mlane = lane & 15;
  int kh = (lane >> 4) << 4;
  const bf16_t* p = tile + (size_t)mlane * CCH;
  v8bf lo = *(const v8bf*)(p + kh);
  v8bf hi = *(const v8bf*)(p + 8 + kh);
  return __builtin_shufflevector(lo, hi, 0, 1, 2, 3, 4, 5, 6, 7,
                                 8, 9, 10, 11, 12, 13, 14, 15);
}

// B-tile loader: 32x16 bf16 from an n-major (n, K) source with ld 256.
// lane<16: col n=lane, K=0..15 contiguous; lane>=16: col n=lane-16, K=16..31.
__device__ __forceinline__ v16bf load_b32x16(const bf16_t* tile, int lane) {
  int nlane = lane & 15;
  int kb = (lane >> 4) << 4;
  return *(const v16bf*)(tile + (size_t)nlane * CCH + kb);
}

// ---------------------------------------------------------------------------
__global__ __launch_bounds__(256) void cvt_bf16_kernel(const float* __restrict__ src,
                                                       bf16_t* __restrict__ dst, int n) {
  int i = blockIdx.x * 256 + threadIdx.x;
  if (i < n) dst[i] = (bf16_t)src[i];
}

// LDS-tiled transpose: x (B, C, HW) f32 -> xT (B, HW, C) bf16.
// 64x64 tiles; coalesced f32 reads, 32B-contiguous bf16 writes.
__global__ __launch_bounds__(256) void transpose_x_kernel(const float* __restrict__ x,
                                                          bf16_t* __restrict__ xT) {
  __shared__ float tile[64][65];
  int ht = blockIdx.x, ct = blockIdx.y, b = blockIdx.z;
  int tid = threadIdx.x;
  const float* src = x + ((size_t)b * CCH + ct * 64) * HWSZ + ht * 64;
  int lc = tid >> 2;
  int lh0 = (tid & 3) * 16;
#pragma unroll
  for (int i = 0; i < 16; ++i) tile[lc][lh0 + i] = src[(size_t)lc * HWSZ + lh0 + i];
  __syncthreads();
  int lh = tid >> 2;
  int lc0 = (tid & 3) * 16;
  bf16_t* dst = xT + ((size_t)b * HWSZ + ht * 64 + lh) * CCH + ct * 64 + lc0;
#pragma unroll
  for (int i = 0; i < 16; ++i) dst[i] = (bf16_t)tile[lc0 + i][lh];
}

// ---------------------------------------------------------------------------
// Generic K=256 bf16 GEMM: out[b, o, n] = sum_c A[o,c] * Bnk[b, n, c].
// One wave per 16x64 output strip: the full A strip (8 x v16bf) stays in
// VGPRs and is reused across 4 n-tiles; B streams with prefetch.
__global__ __launch_bounds__(32) void gemm_proj_kernel(
    const bf16_t* __restrict__ A, const bf16_t* __restrict__ Bnk,
    float* __restrict__ outF,       // (b, 256, Ncols) f32   (optional)
    bf16_t* __restrict__ outT,      // (b, Ncols, 256) bf16  (optional)
    bf16_t* __restrict__ outN,      // (b, 256, Ncols) bf16  (optional)
    int Ncols) {
  int lane = threadIdx.x;
  int ntile0 = blockIdx.x * 4, otile = blockIdx.y, bb = blockIdx.z;
  const bf16_t* Arow = A + (size_t)otile * 16 * CCH;
  v16bf av[8];
#pragma unroll
  for (int kk = 0; kk < 8; ++kk) av[kk] = load_a16x32(Arow + kk * 32, lane);
  int mlane = lane & 15;
  int mofs = (lane >> 4) << 3;
#pragma unroll
  for (int t = 0; t < 4; ++t) {
    const bf16_t* Bbase = Bnk + ((size_t)bb * Ncols + (size_t)(ntile0 + t) * 16) * CCH;
    __builtin_prefetch(Bbase + 16 * CCH, 0, 1);   // next n-tile -> global_prefetch_b8
    v8f acc = {};
#pragma unroll
    for (int kk = 0; kk < 8; ++kk) {
      v16bf bv = load_b32x16(Bbase + kk * 32, lane);
      acc = __builtin_amdgcn_wmma_f32_16x16x32_bf16(false, av[kk], false, bv,
                                                    (short)0, acc, false, false);
    }
#pragma unroll
    for (int vv = 0; vv < 8; ++vv) {
      int m   = otile * 16 + mofs + vv;
      int col = (ntile0 + t) * 16 + mlane;
      float val = acc[vv];
      if (outF) outF[((size_t)bb * CCH + m) * Ncols + col] = val;
      if (outT) outT[((size_t)bb * Ncols + col) * CCH + m] = (bf16_t)val;
      if (outN) outN[((size_t)bb * CCH + m) * Ncols + col] = (bf16_t)val;
    }
  }
}

// ---------------------------------------------------------------------------
// conv_offset: depthwise 5x5 s4 p2 -> BiasFree LN over channels -> GELU ->
// 1x1 to 2ch -> tanh * (1/16) + reference points  => pos (16, 256, 2)
__global__ __launch_bounds__(256) void conv_offset_kernel(
    const float* __restrict__ qf, const float* __restrict__ wdw,
    const float* __restrict__ lnw, const float* __restrict__ wpw,
    float* __restrict__ posb) {
  __shared__ float soff[CG * 256];     // 64 KB of the 320 KB LDS
  int bg = blockIdx.x;
  int b = bg >> 2, g = bg & 3;
  int t = threadIdx.x;
  int py = t >> 4, px = t & 15;
  int y0 = py * 4 - 2, x0 = px * 4 - 2;
  float s1 = 0.f, s2 = 0.f;
  for (int cg = 0; cg < CG; ++cg) {
    const float* img = qf + (size_t)(b * CCH + g * CG + cg) * HWSZ;
    const float* w5  = wdw + cg * 25;
    float acc = 0.f;
#pragma unroll
    for (int ky = 0; ky < 5; ++ky) {
      int yy = y0 + ky;
      if (yy < 0 || yy > HH - 1) continue;
#pragma unroll
      for (int kx = 0; kx < 5; ++kx) {
        int xx = x0 + kx;
        if (xx < 0 || xx > WW - 1) continue;
        acc += w5[ky * 5 + kx] * img[yy * WW + xx];
      }
    }
    soff[cg * 256 + t] = acc;
    s1 += acc; s2 += acc * acc;
  }
  float mu   = s1 * (1.0f / CG);
  float var  = s2 * (1.0f / CG) - mu * mu;    // biased var over channels
  float rstd = rsqrtf(var + 1e-5f);
  float oy = 0.f, ox = 0.f;
  for (int cg = 0; cg < CG; ++cg) {
    float vn = soff[cg * 256 + t] * rstd * lnw[cg];            // BiasFree LN
    float ge = 0.5f * vn * (1.0f + erff(vn * 0.70710678118f)); // exact GELU
    oy += wpw[cg] * ge;
    ox += wpw[CG + cg] * ge;
  }
  float ry = ((py + 0.5f) * (1.0f / 16.0f)) * 2.0f - 1.0f;
  float rx = ((px + 0.5f) * (1.0f / 16.0f)) * 2.0f - 1.0f;
  int n = py * 16 + px;
  posb[(bg * NSAMP + n) * 2 + 0] = tanhf(oy) * (1.0f / 16.0f) + ry;
  posb[(bg * NSAMP + n) * 2 + 1] = tanhf(ox) * (1.0f / 16.0f) + rx;
}

// ---------------------------------------------------------------------------
// grid_sample x at pos -> x_sampled transposed bf16 (b, n, C)
__global__ __launch_bounds__(256) void sample_kernel(
    const float* __restrict__ x, const float* __restrict__ posb,
    bf16_t* __restrict__ xsT) {
  int bg = blockIdx.x, n = threadIdx.x;
  int b = bg >> 2, g = bg & 3;
  float py = posb[(bg * NSAMP + n) * 2 + 0];
  float px = posb[(bg * NSAMP + n) * 2 + 1];
  float xi = (px + 1.0f) * 0.5f * (WW - 1);   // align_corners=True
  float yi = (py + 1.0f) * 0.5f * (HH - 1);
  float xf = floorf(xi), yf = floorf(yi);
  int ix0 = (int)xf, iy0 = (int)yf;
  float wx1 = xi - xf, wy1 = yi - yf;
  float vx0 = (ix0 >= 0 && ix0 <= WW - 1) ? 1.f : 0.f;
  float vx1 = (ix0 + 1 >= 0 && ix0 + 1 <= WW - 1) ? 1.f : 0.f;
  float vy0 = (iy0 >= 0 && iy0 <= HH - 1) ? 1.f : 0.f;
  float vy1 = (iy0 + 1 >= 0 && iy0 + 1 <= HH - 1) ? 1.f : 0.f;
  int cx0 = min(max(ix0, 0), WW - 1), cx1 = min(max(ix0 + 1, 0), WW - 1);
  int cy0 = min(max(iy0, 0), HH - 1), cy1 = min(max(iy0 + 1, 0), HH - 1);
  float m00 = (1 - wy1) * (1 - wx1) * vy0 * vx0;
  float m01 = (1 - wy1) * wx1       * vy0 * vx1;
  float m10 = wy1 * (1 - wx1)       * vy1 * vx0;
  float m11 = wy1 * wx1             * vy1 * vx1;
  for (int cg = 0; cg < CG; ++cg) {
    const float* img = x + (size_t)(b * CCH + g * CG + cg) * HWSZ;
    float val = m00 * img[cy0 * WW + cx0] + m01 * img[cy0 * WW + cx1]
              + m10 * img[cy1 * WW + cx0] + m11 * img[cy1 * WW + cx1];
    xsT[((size_t)b * NSAMP + n) * CCH + g * CG + cg] = (bf16_t)val;
  }
}

// ---------------------------------------------------------------------------
// Fused attention: QK WMMA + RPE bilinear bias + LDS softmax + AV WMMA.
// Block = 4 waves; one (b, head, 16-query tile) per block. attn never hits HBM.
__global__ __launch_bounds__(128) void attn_fused_kernel(
    const bf16_t* __restrict__ qT,   // (b, HW, C)
    const bf16_t* __restrict__ kT,   // (b, n, C)
    const bf16_t* __restrict__ vb,   // (b, C, n)
    const float* __restrict__ posb,  // (bg, n, 2)
    const float* __restrict__ rpe,   // (NH, 127, 127)
    bf16_t* __restrict__ aoT) {      // (b, HW, C)
  __shared__ float sattn[16 * 272];
  int b = blockIdx.z, h = blockIdx.y, mt = blockIdx.x;
  int wave = threadIdx.x >> 5, lane = threadIdx.x & 31;
  int m0 = mt * 16;
  int mlane = lane & 15;
  int kh   = (lane >> 4) << 3;
  int kb16 = (lane >> 4) << 4;
  int mofs = (lane >> 4) << 3;
  const float scale = 0.17677669529663687f;   // 32^-0.5

  // A tile: q^T rows m0..m0+15, channels h*32..h*32+31  (K = hc = 32)
  v16bf aq = load_a16x32(qT + ((size_t)b * HWSZ + m0) * CCH + h * HC, lane);
  int bg = b * NGRP + (h >> 1);
  const float* rpt = rpe + (size_t)h * RPEW * RPEW;

  for (int nt = wave; nt < 16; nt += 4) {
    v16bf bk = *(const v16bf*)(kT + ((size_t)b * NSAMP + nt * 16 + mlane) * CCH
                               + h * HC + kb16);
    v8f acc = {};
    acc = __builtin_amdgcn_wmma_f32_16x16x32_bf16(false, aq, false, bk,
                                                  (short)0, acc, false, false);
    int n = nt * 16 + mlane;
    float py = posb[(bg * NSAMP + n) * 2 + 0];
    float px = posb[(bg * NSAMP + n) * 2 + 1];
#pragma unroll
    for (int vv = 0; vv < 8; ++vv) {
      int m  = m0 + mofs + vv;
      int iy = m >> 6, ix = m & 63;
      float gy = ((iy + 0.5f) * (1.0f / 64.0f)) * 2.0f - 1.0f;
      float gx = ((ix + 0.5f) * (1.0f / 64.0f)) * 2.0f - 1.0f;
      float dy = (gy - py) * 0.5f;
      float dx = (gx - px) * 0.5f;
      float xi = (dx + 1.0f) * 0.5f * (RPEW - 1);   // align_corners=True
      float yi = (dy + 1.0f) * 0.5f * (RPEW - 1);
      float xf = floorf(xi), yf = floorf(yi);
      int ix0 = (int)xf, iy0 = (int)yf;
      float wx1 = xi - xf, wy1 = yi - yf;
      float bias = 0.f;
#pragma unroll
      for (int ty = 0; ty < 2; ++ty) {
#pragma unroll
        for (int tx = 0; tx < 2; ++tx) {
          int yy = iy0 + ty, xx = ix0 + tx;
          float w = (ty ? wy1 : 1.0f - wy1) * (tx ? wx1 : 1.0f - wx1);
          if (yy >= 0 && yy < RPEW && xx >= 0 && xx < RPEW)
            bias += w * rpt[yy * RPEW + xx];
        }
      }
      sattn[(mofs + vv) * 272 + n] = acc[vv] * scale + bias;
    }
  }
  __syncthreads();

  // Softmax over 256 samples, one row per thread
  if (threadIdx.x < 16) {
    float* row = &sattn[threadIdx.x * 272];
    float mx = -1e30f;
    for (int j = 0; j < NSAMP; ++j) mx = fmaxf(mx, row[j]);
    float s = 0.f;
    for (int j = 0; j < NSAMP; ++j) { float ev = __expf(row[j] - mx); row[j] = ev; s += ev; }
    float inv = 1.0f / s;
    for (int j = 0; j < NSAMP; ++j) row[j] *= inv;
  }
  __syncthreads();

  // out[m, c] = sum_n attn[m, n] * v[c, n]; K = 256 = 16 WMMA steps; waves 0/1
  if (wave < 2) {
    int c0 = wave * 16;
    v8f acc = {};
    for (int kk = 0; kk < 16; ++kk) {
      v16bf aa;
      const float* arow = &sattn[mlane * 272 + kk * 32];
#pragma unroll
      for (int i = 0; i < 8; ++i) {
        aa[i]     = (bf16_t)arow[kb16 + i];
        aa[8 + i] = (bf16_t)arow[kb16 + 8 + i];
      }
      v16bf bb = *(const v16bf*)(vb + ((size_t)b * CCH + h * HC + c0 + mlane) * NSAMP
                                 + kk * 32 + kb16);
      acc = __builtin_amdgcn_wmma_f32_16x16x32_bf16(false, aa, false, bb,
                                                    (short)0, acc, false, false);
    }
#pragma unroll
    for (int vv = 0; vv < 8; ++vv)
      aoT[((size_t)b * HWSZ + m0 + mofs + vv) * CCH + h * HC + c0 + mlane] = (bf16_t)acc[vv];
  }
}

// ---------------------------------------------------------------------------
extern "C" void kernel_launch(void* const* d_in, const int* in_sizes, int n_in,
                              void* d_out, int out_size, void* d_ws, size_t ws_size,
                              hipStream_t stream) {
  const float* x     = (const float*)d_in[0];
  const float* w_q   = (const float*)d_in[1];
  const float* w_k   = (const float*)d_in[2];
  const float* w_v   = (const float*)d_in[3];
  const float* w_out = (const float*)d_in[4];
  const float* w_dw  = (const float*)d_in[5];
  const float* ln_w  = (const float*)d_in[6];
  const float* w_pw  = (const float*)d_in[7];
  const float* rpe   = (const float*)d_in[8];
  float* out = (float*)d_out;

  char* ws = (char*)d_ws;
  size_t off = 0;
  auto take = [&](size_t bytes) -> char* {
    char* p = ws + off;
    off += (bytes + 255) & ~(size_t)255;
    return p;
  };
  bf16_t* wq_bf = (bf16_t*)take((size_t)CCH * CCH * 2);
  bf16_t* wk_bf = (bf16_t*)take((size_t)CCH * CCH * 2);
  bf16_t* wv_bf = (bf16_t*)take((size_t)CCH * CCH * 2);
  bf16_t* wo_bf = (bf16_t*)take((size_t)CCH * CCH * 2);
  bf16_t* xT    = (bf16_t*)take((size_t)BATCH * HWSZ * CCH * 2);
  float*  qf    = (float*) take((size_t)BATCH * CCH * HWSZ * 4);
  bf16_t* qT    = (bf16_t*)take((size_t)BATCH * HWSZ * CCH * 2);
  float*  posb  = (float*) take((size_t)BATCH * NGRP * NSAMP * 2 * 4);
  bf16_t* xsT   = (bf16_t*)take((size_t)BATCH * NSAMP * CCH * 2);
  bf16_t* kT    = (bf16_t*)take((size_t)BATCH * NSAMP * CCH * 2);
  bf16_t* vb    = (bf16_t*)take((size_t)BATCH * CCH * NSAMP * 2);
  bf16_t* aoT   = (bf16_t*)take((size_t)BATCH * HWSZ * CCH * 2);
  (void)ws_size; (void)in_sizes; (void)n_in; (void)out_size;

  const int nW = CCH * CCH;
  cvt_bf16_kernel<<<nW / 256, 256, 0, stream>>>(w_q,   wq_bf, nW);
  cvt_bf16_kernel<<<nW / 256, 256, 0, stream>>>(w_k,   wk_bf, nW);
  cvt_bf16_kernel<<<nW / 256, 256, 0, stream>>>(w_v,   wv_bf, nW);
  cvt_bf16_kernel<<<nW / 256, 256, 0, stream>>>(w_out, wo_bf, nW);

  transpose_x_kernel<<<dim3(HWSZ / 64, CCH / 64, BATCH), 256, 0, stream>>>(x, xT);

  // q = w_q @ x  -> q f32 (b,c,hw) for conv_offset, q^T bf16 for attention
  gemm_proj_kernel<<<dim3(HWSZ / 64, CCH / 16, BATCH), 32, 0, stream>>>(
      wq_bf, xT, qf, qT, (bf16_t*)nullptr, HWSZ);

  conv_offset_kernel<<<BATCH * NGRP, 256, 0, stream>>>(qf, w_dw, ln_w, w_pw, posb);

  sample_kernel<<<BATCH * NGRP, 256, 0, stream>>>(x, posb, xsT);

  // k^T (b,n,C) and v (b,C,n), both bf16
  gemm_proj_kernel<<<dim3(NSAMP / 64, CCH / 16, BATCH), 32, 0, stream>>>(
      wk_bf, xsT, (float*)nullptr, kT, (bf16_t*)nullptr, NSAMP);
  gemm_proj_kernel<<<dim3(NSAMP / 64, CCH / 16, BATCH), 32, 0, stream>>>(
      wv_bf, xsT, (float*)nullptr, (bf16_t*)nullptr, vb, NSAMP);

  attn_fused_kernel<<<dim3(HWSZ / 16, NHEAD, BATCH), 128, 0, stream>>>(
      qT, kT, vb, posb, rpe, aoT);

  // final projection -> d_out f32 (B, C, H, W)
  gemm_proj_kernel<<<dim3(HWSZ / 64, CCH / 16, BATCH), 32, 0, stream>>>(
      wo_bf, aoT, out, (bf16_t*)nullptr, (bf16_t*)nullptr, HWSZ);
}
